// CWRNN_88132728914569
// MI455X (gfx1250) — matrix operand order repaired and checked by reading, not compiled, not run on hardware
//
#include <hip/hip_runtime.h>
#include <hip/hip_bf16.h>

// Clockwork RNN on MI455X (gfx1250):
//  - batch/head-parallel recurrence, full T-loop inside one kernel (no grid sync needed)
//  - per-head weights staged once into 320KB-class LDS as bf16, transposed [n][k]
//  - V_WMMA_F32_16X16X32_BF16 with fp32 accumulate, hardware V_TANH_F32 per step
//  - time loop unrolled by clockwork period 4 -> tile counts are compile-time (no EXEC churn)
//  - k-loop software-pipelined: next k-step's ds_load_b128s issued before current WMMA pair

typedef __attribute__((ext_vector_type(16))) __bf16 v16bf;
typedef __attribute__((ext_vector_type(8)))  float  v8f;

#define B_SZ    512
#define T_STEPS 192
#define DXF     32
#define DYH     4
#define DH1     256
#define DH2     512
#define KTOT    288      // DH1 + DXF
#define KP      296      // padded LDS k-stride: mult of 8 (16B rows), bank-conflict-free
#define MT      16       // batch rows per workgroup (WMMA M)

#if __has_builtin(__builtin_amdgcn_tanhf)
#define TANHF(x) __builtin_amdgcn_tanhf(x)     // V_TANH_F32 trans op
#else
#define TANHF(x) tanhf(x)
#endif

union OpU { uint4 q[2]; v16bf v; };

#define LDOP(dst, p, off) do {                                   \
    (dst).q[0] = *(const uint4*)((p) + (off));                   \
    (dst).q[1] = *(const uint4*)((p) + (off) + 16); } while (0)

#define WMMA_BF16(Aop, Bop, Cacc)                                \
    __builtin_amdgcn_wmma_f32_16x16x32_bf16(false, (Aop).v, false, (Bop).v, (short)0, (Cacc), false, false)

// One clockwork step. NTL in {8,12,16} tiles of 16 output cols; NFN = #features for step tt+1.
#define STEP(tt, NTL, NFN) do {                                                        \
    bool t1on;                                                                         \
    if ((NTL) == 16)      t1on = true;                                                 \
    else if ((NTL) == 8)  t1on = false;                                                \
    else t1on = __builtin_amdgcn_readfirstlane((wid < 4) ? 1 : 0) != 0;                \
    v8f acc0 = vzero, acc1 = vzero;                                                    \
    if (t1on) {                                                                        \
        OpU Ax[2], Bx[2], Cx[2];                                                       \
        LDOP(Ax[0], pA0, 0); LDOP(Bx[0], pB0, 0); LDOP(Cx[0], pB1, 0);                 \
        _Pragma("unroll")                                                              \
        for (int i = 0; i < 9; ++i) {                                                  \
            const int cu = i & 1, nx = cu ^ 1;                                         \
            if (i < 8) {                                                               \
                LDOP(Ax[nx], pA0, (i+1)*32);                                           \
                LDOP(Bx[nx], pB0, (i+1)*32);                                           \
                LDOP(Cx[nx], pB1, (i+1)*32);                                           \
            }                                                                          \
            acc0 = WMMA_BF16(Ax[cu], Bx[cu], acc0);                                    \
            acc1 = WMMA_BF16(Ax[cu], Cx[cu], acc1);                                    \
        }                                                                              \
    } else {                                                                           \
        OpU Ax[2], Bx[2];                                                              \
        LDOP(Ax[0], pA0, 0); LDOP(Bx[0], pB0, 0);                                      \
        _Pragma("unroll")                                                              \
        for (int i = 0; i < 9; ++i) {                                                  \
            const int cu = i & 1, nx = cu ^ 1;                                         \
            if (i < 8) {                                                               \
                LDOP(Ax[nx], pA0, (i+1)*32);                                           \
                LDOP(Bx[nx], pB0, (i+1)*32);                                           \
            }                                                                          \
            acc0 = WMMA_BF16(Ax[cu], Bx[cu], acc0);                                    \
        }                                                                              \
    }                                                                                  \
    __syncthreads();  /* all LDS reads of h_t complete before masked update */         \
    _Pragma("unroll")                                                                  \
    for (int v = 0; v < 8; ++v) st0[v*KP] = (__bf16)TANHF(acc0[v] + bias0);            \
    if (t1on) {                                                                        \
        _Pragma("unroll")                                                              \
        for (int v = 0; v < 8; ++v) st1[v*KP] = (__bf16)TANHF(acc1[v] + bias1);        \
    }                                                                                  \
    if ((tt) + 1 < T_STEPS) {  /* stage masked x for step tt+1 */                      \
        float xv0 = 0.f, xv1 = 0.f;                                                    \
        if (((NFN) == 32) || (f0 < (NFN))) {                                           \
            xv0 = xp0[((tt)+1)*DXF];                                                   \
            xv1 = xp1[((tt)+1)*DXF];                                                   \
        }                                                                              \
        *hx0 = (__bf16)xv0; *hx1 = (__bf16)xv1;                                        \
    }                                                                                  \
    __syncthreads();                                                                   \
} while (0)

extern "C" __global__ __launch_bounds__(256)
void cwrnn_scan(const float* __restrict__ x,
                const float* __restrict__ w_x,
                const float* __restrict__ w_h,
                const float* __restrict__ bvec,
                const float* __restrict__ W1,
                const float* __restrict__ b1,
                const float* __restrict__ W2,
                const float* __restrict__ b2,
                float* __restrict__ out)
{
    extern __shared__ __align__(16) char smem[];
    __bf16* Waug = (__bf16*)smem;                      // [DH1][KP]  transposed [n][k]
    __bf16* hA   = (__bf16*)(smem + DH1 * KP * 2);     // [MT][KP]   [row][k], k>=256 = masked x

    const int tid   = threadIdx.x;
    const int lane  = tid & 31;
    const int wid   = tid >> 5;          // 8 waves
    const int ytile = blockIdx.y;        // head 0..3
    const int bbase = blockIdx.x * MT;   // batch base

    // ---- stage augmented weights into LDS (bf16, transposed), coalesced global reads ----
    for (int idx = tid; idx < KTOT * DH1; idx += 256) {
        int kd = idx >> 8;               // contraction index 0..287
        int n  = idx & (DH1 - 1);        // output column (consecutive -> coalesced)
        float w = (kd < DH1) ? w_h[((ytile * DH1) + kd) * DH1 + n]
                             : w_x[((ytile * DXF) + (kd - DH1)) * DH1 + n];
        Waug[n * KP + kd] = (__bf16)w;
    }
    // ---- h0 = 0 ----
    for (int idx = tid; idx < MT * DH1; idx += 256) {
        int r = idx >> 8, d = idx & (DH1 - 1);
        hA[r * KP + d] = (__bf16)0.f;
    }

    // ---- per-thread x staging slots (each thread owns 2 of the 16x32 x-tile) ----
    const int f0 = tid & (DXF - 1);
    const int r0 = tid >> 5;             // 0..7
    const int r1 = r0 + 8;               // 8..15
    const float* xp0 = x + ((size_t)(bbase + r0) * T_STEPS) * DXF + f0;
    const float* xp1 = x + ((size_t)(bbase + r1) * T_STEPS) * DXF + f0;
    __bf16* hx0 = hA + r0 * KP + DH1 + f0;
    __bf16* hx1 = hA + r1 * KP + DH1 + f0;
    {   // stage masked x for step 0 (nfeat(0) = 16)
        float xv0 = 0.f, xv1 = 0.f;
        if (f0 < 16) { xv0 = xp0[0]; xv1 = xp1[0]; }
        *hx0 = (__bf16)xv0; *hx1 = (__bf16)xv1;
    }
    __syncthreads();

    // ---- per-wave WMMA operand addressing (ISA 7.12.2 layouts) ----
    const int n0 = wid * 16 + (lane & 15);          // tile0 output column
    const int n1 = (wid + 8) * 16 + (lane & 15);    // tile1 output column
    const float bias0 = bvec[ytile * DH1 + n0];
    const float bias1 = bvec[ytile * DH1 + n1];
    const int rowA  = lane & 15;
    const int koffA = (lane >> 4) << 3;             // A: upper half-wave holds K+8 halves
    const int koffB = (lane >> 4) << 3;             // B: upper half-wave holds K+8 halves
    const int crow  = (lane >> 4) << 3;             // C: upper half-wave holds rows 8..15
    const __bf16* pA0 = hA   + rowA * KP + koffA;
    const __bf16* pB0 = Waug + n0   * KP + koffB;
    const __bf16* pB1 = Waug + n1   * KP + koffB;
    __bf16* st0 = hA + crow * KP + n0;              // cand writeback bases
    __bf16* st1 = hA + crow * KP + n1;
    const v8f vzero = {0.f, 0.f, 0.f, 0.f, 0.f, 0.f, 0.f, 0.f};

    // ---- clockwork scan, unrolled by schedule period 4:
    //      ntiles(t%4) = {8,12,8,16}; nfeat(t+1) = {24,16,32,16} ----
    #pragma unroll 1
    for (int t = 0; t < T_STEPS; t += 4) {
        STEP(t + 0,  8, 24);
        STEP(t + 1, 12, 16);
        STEP(t + 2,  8, 32);
        STEP(t + 3, 16, 16);
    }

    // ---- head: hid = relu(hT @ W1 + b1)  (runs once; L2-resident W1, coalesced) ----
    float* hid = (float*)smem;          // reuse weight region: 16*512 f32 = 32KB
    for (int idx = tid; idx < MT * DH2; idx += 256) {
        int r = idx >> 9, c = idx & (DH2 - 1);
        float s = b1[ytile * DH2 + c];
        const float* w1p = W1 + (size_t)(ytile * DH1) * DH2 + c;
        #pragma unroll 4
        for (int k = 0; k < DH1; ++k)
            s += (float)hA[r * KP + k] * w1p[(size_t)k * DH2];
        hid[idx] = fmaxf(s, 0.f);
    }
    __syncthreads();

    // ---- y = hid @ W2 + b2, tree-reduced through LDS ----
    float* part = hid + MT * DH2;       // [16 parts][16 rows]
    {
        int r = tid & 15, p = tid >> 4;
        float s = 0.f;
        for (int c = p * 32; c < p * 32 + 32; ++c)
            s += hid[r * DH2 + c] * W2[ytile * DH2 + c];
        part[p * 16 + r] = s;
    }
    __syncthreads();
    if (tid < MT) {
        float s = b2[ytile];
        #pragma unroll
        for (int p = 0; p < 16; ++p) s += part[p * 16 + tid];
        out[(bbase + tid) * DYH + ytile] = s;
    }
}

extern "C" void kernel_launch(void* const* d_in, const int* in_sizes, int n_in,
                              void* d_out, int out_size, void* d_ws, size_t ws_size,
                              hipStream_t stream) {
    (void)in_sizes; (void)n_in; (void)d_ws; (void)ws_size; (void)out_size;
    const float* x   = (const float*)d_in[0];
    const float* w_x = (const float*)d_in[1];
    const float* w_h = (const float*)d_in[2];
    const float* b   = (const float*)d_in[3];
    const float* W1  = (const float*)d_in[4];
    const float* b1  = (const float*)d_in[5];
    const float* W2  = (const float*)d_in[6];
    const float* b2  = (const float*)d_in[7];
    float* out = (float*)d_out;

    dim3 grid(B_SZ / MT, DYH);                       // 32 batch tiles x 4 heads = 128 WGs
    size_t lds = (size_t)(DH1 + MT) * KP * 2;        // 161,024 B
    cwrnn_scan<<<grid, dim3(256), lds, stream>>>(x, w_x, w_h, b, W1, b1, W2, b2, out);
}
